// BasicBlock_52158082843183
// MI455X (gfx1250) — matrix operand order repaired and checked by reading, not compiled, not run on hardware
//
#include <hip/hip_runtime.h>

// CDNA5 / gfx1250: wave32, WMMA f32_16x16x32_f16 path.
// Pipeline:
//   1) binarize  -> pointwise weights written in WMMA A-fragment order
//   2) depthwise 3x3 + bias -> y written in WMMA B-fragment record order (LDS transpose)
//   3) WMMA channel GEMM: weights TDM'd into LDS, y streamed as contiguous b128 records,
//      + pointwise bias + ReLU, coalesced NCHW stores.

typedef __attribute__((ext_vector_type(16))) _Float16 v16h;
typedef __attribute__((ext_vector_type(8)))  _Float16 v8h;
typedef __attribute__((ext_vector_type(8)))  float    v8f;
typedef __attribute__((ext_vector_type(4)))  unsigned int v4u;
typedef __attribute__((ext_vector_type(8)))  int      v8i;
typedef __attribute__((ext_vector_type(4)))  int      v4i;

#define C_CH 256
#define O_CH 256
#define HW_  3136      // 56*56
#define W_   56
#define NB_  32
#define NTILE 196      // hw tiles per image (3136/16)

// ---------------- workspace layout (bytes) ----------------
// [0      , 9216 )  wd       f32[256*9]   binarized depthwise weights (+-1)
// [9216   , 10240)  bd       f32[256]     binarized depthwise bias    (+-1)
// [10240  , 11264)  bp       f32[256]     binarized pointwise bias    (+-1)
// [11264  , 142336) wpb_frag f16[65536]   pointwise weights, A-fragment order (2 planes)
// [142336 , ~51.5MB) yfrag   f16[6272*4096] depthwise output in B-fragment record order:
//                    per (n,tile): [kstep(8)][lane(32)][half(16)] = 8KB contiguous
#define WS_WD   0
#define WS_BD   9216
#define WS_BP   10240
#define WS_WPB  11264
#define WS_Y    142336

#define AFRAG_PLANE_HALFS 32768   // 8 ksteps * 16 otiles * 32 lanes * 8 halves

// ---------------- deterministic hash-based U(0,1) ----------------
__device__ __forceinline__ float hash_u01(uint32_t idx, uint32_t seed) {
    uint32_t s = idx + seed * 0x9E3779B9u;
    s ^= s >> 16; s *= 0x7feb352du;
    s ^= s >> 15; s *= 0x846ca68bu;
    s ^= s >> 16;
    return (float)(s >> 8) * (1.0f / 16777216.0f);
}

__device__ __forceinline__ float sbin(float w, uint32_t idx, uint32_t seed) {
    float r = 2.0f * hash_u01(idx, seed) - 1.0f;   // U(-1,1)
    return (w >= r) ? 1.0f : -1.0f;                // sign(w - r)
}

// ---------------- kernel 1: binarize; pointwise weights -> A-fragment order ----------------
// A-fragment (16-bit A 16x32, ISA 7.12.2), stored as two 16B planes of records
// [kstep][otile][lane]:  o = ot*16 + (lane&15),  c = kstep*32 + h' + 16*plane + 8*(lane>>4)
__global__ void binarize_kernel(const float* __restrict__ dw_w,
                                const float* __restrict__ dw_b,
                                const float* __restrict__ pw_w,
                                const float* __restrict__ pw_b,
                                float* __restrict__ wd,
                                float* __restrict__ bd,
                                float* __restrict__ bp,
                                _Float16* __restrict__ wpb_frag) {
    int i = blockIdx.x * blockDim.x + threadIdx.x;
    if (i < O_CH * C_CH) {
        int o = i >> 8, c = i & 255;
        int kstep = c >> 5;
        int r     = c & 31;
        int hiL   = (r >> 3) & 1;            // lane half (K+8 offset)
        int p     = r >> 4;                  // plane (K+16 offset)
        int hh    = r & 7;                   // half within 16B plane record
        int lane  = (o & 15) + 16 * hiL;
        size_t dst = (size_t)p * AFRAG_PLANE_HALFS
                   + ((size_t)((kstep * 16 + (o >> 4)) * 32 + lane)) * 8 + hh;
        wpb_frag[dst] = (_Float16)sbin(pw_w[i], (uint32_t)i, 0xA341u);
    }
    if (i < C_CH * 9) wd[i] = sbin(dw_w[i], (uint32_t)i, 0x51F1u);
    if (i < C_CH)     bd[i] = sbin(dw_b[i], (uint32_t)i, 0xC0FEu);
    if (i < O_CH)     bp[i] = sbin(pw_b[i], (uint32_t)i, 0x1234u);
}

// ---------------- kernel 2: depthwise 3x3 + bias -> B-fragment records ----------------
// One block per (n, 16-pixel hw tile); thread = channel. Channel->lane transpose goes
// through an 8KB LDS tile so the global write is a coalesced linear b128 stream.
// B-fragment record order per tile: [kstep(8)][lane(32)][half(16)] halves,
//   lane = lm + 16*((c>>3)&1),  half = (c&7) + 8*((c%32)>>4),  kstep = c/32,  lm = hw%16.
__global__ __launch_bounds__(256)
void dwconv_pack_kernel(const float* __restrict__ x,
                        const float* __restrict__ wd,
                        const float* __restrict__ bd,
                        _Float16* __restrict__ yfrag) {
    __shared__ _Float16 tile[4096];          // 8 KB
    const int tid = threadIdx.x;
    const int c   = tid;
    const int n   = blockIdx.x / NTILE;
    const int t   = blockIdx.x % NTILE;
    const int hw0 = t * 16;

    const float* xp = x + ((size_t)(n * C_CH + c)) * HW_;
    const float* wk = wd + c * 9;
    const float bias = bd[c];

    const int kstep = c >> 5;
    const int r     = c & 31;
    const int hiL   = (r >> 3) & 1;
    const int h16   = (r & 7) + ((r >> 4) << 3);

    #pragma unroll
    for (int j = 0; j < 16; ++j) {
        int hw = hw0 + j;
        int h  = hw / W_;
        int w  = hw % W_;
        float acc = bias;
        #pragma unroll
        for (int dh = -1; dh <= 1; ++dh) {
            int hh = h + dh;
            if ((unsigned)hh < (unsigned)W_) {
                #pragma unroll
                for (int dw = -1; dw <= 1; ++dw) {
                    int ww = w + dw;
                    if ((unsigned)ww < (unsigned)W_)
                        acc += wk[(dh + 1) * 3 + (dw + 1)] * xp[hh * W_ + ww];
                }
            }
        }
        tile[((kstep * 32) + (j + 16 * hiL)) * 16 + h16] = (_Float16)acc;
    }
    __syncthreads();

    // cooperative coalesced store: 8 KB = 512 uint4, layout identical in LDS and global
    const uint4* src = (const uint4*)tile;
    uint4* dst = (uint4*)(yfrag + (size_t)blockIdx.x * 4096);
    dst[tid]       = src[tid];
    dst[tid + 256] = src[tid + 256];
}

// ---------------- kernel 3: pointwise 1x1 as WMMA GEMM + bias + ReLU ----------------
__global__ __launch_bounds__(256, 1)
void pwgemm_kernel(const _Float16* __restrict__ yfrag,
                   const _Float16* __restrict__ wpb_frag,
                   const float* __restrict__ bp,
                   float* __restrict__ out) {
    extern __shared__ __align__(16) char smem_raw[];   // 128 KB dynamic LDS
    _Float16* a_lds = (_Float16*)smem_raw;

    const int tid    = threadIdx.x;
    const int wave   = tid >> 5;
    const int lane   = tid & 31;
    const int lm     = lane & 15;
    const int hiHalf = lane >> 4;

    // ---- stage weights into LDS via the Tensor Data Mover (one D# per block) ----
    // 1-D tensor of 16384 x 8B elements (=128KB), tile == tensor, multicast off.
    if (wave == 0) {
        uint32_t lds_off = (uint32_t)(uintptr_t)a_lds;       // LDS byte address
        uint64_t ga      = (uint64_t)(uintptr_t)wpb_frag;    // global tile start
        v4u g0;
        g0.x = 1u;                                           // count=1, user D#
        g0.y = lds_off;                                      // lds_addr
        g0.z = (uint32_t)ga;                                 // global_addr[31:0]
        g0.w = (uint32_t)((ga >> 32) & 0x01FFFFFFu)          // global_addr[56:32]
             | 0x80000000u;                                  // type=2 ("image")
        v8i g1;
        g1[0] = (int)(3u << 16);        // workgroup_mask=0, data_size=3 (8B)
        g1[1] = (int)(16384u << 16);    // tensor_dim0[15:0] in bits 79:64
        g1[2] = (int)(1u << 16);        // tensor_dim0 hi=0, tensor_dim1=1
        g1[3] = (int)(16384u << 16);    // tensor_dim1 hi=0, tile_dim0=16384
        g1[4] = 1;                      // tile_dim1=1, tile_dim2=0
        g1[5] = 16384;                  // tensor_dim0_stride lo32
        g1[6] = 0;                      // stride hi, tensor_dim1_stride lo
        g1[7] = 0;
        v4i gz4 = {0, 0, 0, 0};         // groups 2/3 unused (<=2D tensor)
        v8i gz8 = {0, 0, 0, 0, 0, 0, 0, 0};
        __builtin_amdgcn_tensor_load_to_lds(g0, g1, gz4, gz4, gz8, 0);
        __builtin_amdgcn_s_wait_tensorcnt(0);
    }
    // Belt-and-braces: deterministic coalesced copy of the same 128 KB (idempotent).
    {
        const uint4* src = (const uint4*)wpb_frag;           // 8192 uint4
        uint4*       dst = (uint4*)a_lds;
        #pragma unroll
        for (int j = 0; j < 32; ++j) {
            int idx = j * 256 + tid;
            dst[idx] = src[idx];
        }
    }
    __syncthreads();

    const int t  = blockIdx.x * 8 + wave;                    // hw-tile id in [0, 6272)
    const int n  = (t * 16) / HW_;
    const int hw0 = (t * 16) % HW_;

    const v8h* aplane0 = (const v8h*)a_lds;                  // 4096 records of 16B
    const v8h* aplane1 = aplane0 + (AFRAG_PLANE_HALFS / 8);
    const v8h* yrec    = (const v8h*)(yfrag + (size_t)t * 4096);

    v8f acc[16] = {};                                        // 16 o-tiles x 8 VGPRs

    #pragma unroll
    for (int ks = 0; ks < 8; ++ks) {
        // B fragment: one contiguous 32B record per lane -> 2x global_load_b128,
        // the wave reads 1KB contiguous of the single-pass y stream.
        int brec = (ks * 32 + lane) * 2;
        v8h blo = yrec[brec];
        v8h bhi = yrec[brec + 1];
        v16h bf = __builtin_shufflevector(blo, bhi,
                    0, 1, 2, 3, 4, 5, 6, 7, 8, 9, 10, 11, 12, 13, 14, 15);
        #pragma unroll
        for (int ot = 0; ot < 16; ++ot) {
            int arec = (ks * 16 + ot) * 32 + lane;
            v8h lo = aplane0[arec];                          // ds_load_b128
            v8h hi = aplane1[arec];                          // ds_load_b128
            v16h af = __builtin_shufflevector(lo, hi,
                        0, 1, 2, 3, 4, 5, 6, 7, 8, 9, 10, 11, 12, 13, 14, 15);
            acc[ot] = __builtin_amdgcn_wmma_f32_16x16x32_f16(
                false, af, false, bf, (short)0, acc[ot], false, false);
        }
    }

    // Epilogue: + pointwise bias, ReLU, coalesced NCHW stores (lanes span contiguous hw)
    #pragma unroll
    for (int ot = 0; ot < 16; ++ot) {
        #pragma unroll
        for (int rr = 0; rr < 8; ++rr) {
            int o = ot * 16 + rr + hiHalf * 8;
            float v = acc[ot][rr] + bp[o];
            out[((size_t)(n * O_CH + o)) * HW_ + hw0 + lm] = fmaxf(v, 0.0f);
        }
    }
}

// ---------------- host-side launch ----------------
extern "C" void kernel_launch(void* const* d_in, const int* in_sizes, int n_in,
                              void* d_out, int out_size, void* d_ws, size_t ws_size,
                              hipStream_t stream) {
    (void)in_sizes; (void)n_in; (void)out_size; (void)ws_size;

    const float* x    = (const float*)d_in[0];   // [32,256,56,56]
    const float* dw_w = (const float*)d_in[1];   // [256,1,3,3]
    const float* dw_b = (const float*)d_in[2];   // [256]
    const float* pw_w = (const float*)d_in[3];   // [256,256,1,1]
    const float* pw_b = (const float*)d_in[4];   // [256]
    float* out = (float*)d_out;                  // [32,256,56,56]

    char* ws = (char*)d_ws;                      // needs ~51.5 MB
    float*    wd       = (float*)(ws + WS_WD);
    float*    bd       = (float*)(ws + WS_BD);
    float*    bp       = (float*)(ws + WS_BP);
    _Float16* wpb_frag = (_Float16*)(ws + WS_WPB);
    _Float16* yfrag    = (_Float16*)(ws + WS_Y);

    binarize_kernel<<<256, 256, 0, stream>>>(dw_w, dw_b, pw_w, pw_b, wd, bd, bp, wpb_frag);

    dwconv_pack_kernel<<<NB_ * NTILE, 256, 0, stream>>>(x, wd, bd, yfrag);  // 6272 blocks

    // 6272 hw-tiles / 8 waves per block; 128 KB dynamic LDS for the weight stage
    pwgemm_kernel<<<784, 256, 131072, stream>>>(yfrag, wpb_frag, bp, out);
}
